// LearnedSimulator_2293512536194
// MI455X (gfx1250) — matrix operand order repaired and checked by reading, not compiled, not run on hardware
//
#include <hip/hip_runtime.h>

#define H   128
#define TM  128      // rows per block
#define TK  32       // K step
#define NT  8        // 16-col tiles per wave (full 128 cols)

typedef __attribute__((ext_vector_type(16))) __bf16 v16bf;
typedef __attribute__((ext_vector_type(8)))  float  v8f;

struct U128 { unsigned int x, y, z, w; };
union Frag16 { v16bf v; U128 q[2]; };

__device__ __forceinline__ unsigned short f32_bf16(float f) {
    unsigned int u = __float_as_uint(f);
    u += 0x7FFFu + ((u >> 16) & 1u);          // round-to-nearest-even
    return (unsigned short)(u >> 16);
}

// ---------------------------------------------------------------------------
// WMMA GEMM:  out[M x 128] = epilogue(A[M x K] @ W[K x 128] + bias)
// A is bf16 everywhere (activations stored bf16; fp32 masters kept separately).
//   MODE 0: A = A0 (contiguous bf16 [M][K])
//   MODE 1: A = cat(nf16[dst[e]], nf16[src[e]], ef16[e])   (K = 384)
//   MODE 2: A = cat(A0[row], A1[row])                      (K = 256)
//   EPI  0: relu -> C16 (bf16)
//   EPI  1: linear -> C (fp32)
//   EPI  2: edge final: C += y (fp32 master); C16 = bf16(C); atomicAdd(C2[dst])
//   EPI  3: node residual: C += y; C16 = bf16(C)
//   EPI  4: linear dual: C = y; C16 = bf16(y)
// 256 threads = 8 waves; wave w owns rows [w*16, w*16+16) x all 128 cols.
// Double-buffered LDS: global loads for tile k+1 issue before tile k's WMMAs.
// ---------------------------------------------------------------------------
template<int MODE, int EPI>
__global__ __launch_bounds__(256)
void gemm128_wmma_bf16(int M, int K,
                       const unsigned short* __restrict__ A0,
                       const unsigned short* __restrict__ A1,
                       const int*   __restrict__ srcIdx,
                       const int*   __restrict__ dstIdx,
                       const unsigned short* __restrict__ WT,   // bf16 [128][K]
                       const float* __restrict__ bias,
                       float* __restrict__ C,
                       float* __restrict__ C2,
                       unsigned short* __restrict__ C16)
{
    __shared__ unsigned short lA[2][TM * TK];   // 2 x 8 KB
    __shared__ unsigned short lB[2][H  * TK];   // 2 x 8 KB

    const int tid  = threadIdx.x;
    const int lane = tid & 31;
    const int wave = tid >> 5;
    const int half = lane >> 4;          // wave32 half
    const int l16  = lane & 15;
    const int m0   = blockIdx.x * TM;
    const int wm   = wave * 16;          // wave's row tile

    // staging: each thread copies 32 bytes of A and 32 bytes of B per K-step
    const int rA = tid >> 1;             // 0..127 (A row / B col index)
    const int kA = (tid & 1) * 16;       // 0 or 16 shorts

    // hoist per-row base pointers (gather indices read once, not per K-step)
    const int rowS = m0 + rA;
    const bool rowOK = rowS < M;
    const unsigned short *pR0 = nullptr, *pR1 = nullptr, *pR2 = nullptr;
    if (rowOK) {
        if (MODE == 0) {
            pR0 = A0 + (size_t)rowS * K;
        } else if (MODE == 1) {
            pR0 = A0 + (size_t)dstIdx[rowS] * H;
            pR1 = A0 + (size_t)srcIdx[rowS] * H;
            pR2 = A1 + (size_t)rowS * H;
        } else {
            pR0 = A0 + (size_t)rowS * H;
            pR1 = A1 + (size_t)rowS * H;
        }
    }
    const unsigned short* pB = WT + (size_t)rA * K;

    v8f acc[NT] = {};
    U128 ra0, ra1, rb0, rb1;

    auto loadTile = [&](int k0) {
        const int kg = k0 + kA;          // 16-aligned; never crosses a 128-boundary
        const unsigned short* pa = nullptr;
        if (rowOK) {
            if (MODE == 0)      pa = pR0 + kg;
            else if (MODE == 1) pa = (kg < H) ? pR0 + kg
                                   : (kg < 2 * H) ? pR1 + (kg - H)
                                   : pR2 + (kg - 2 * H);
            else                pa = (kg < H) ? pR0 + kg : pR1 + (kg - H);
        }
        if (pa) {
            ra0 = *reinterpret_cast<const U128*>(pa);
            ra1 = *reinterpret_cast<const U128*>(pa + 8);
            if (MODE == 0 && k0 + TK < K)
                __builtin_prefetch(pa + TK, 0, 0);      // global_prefetch_b8
        } else {
            ra0 = U128{0, 0, 0, 0}; ra1 = U128{0, 0, 0, 0};
        }
        const unsigned short* pb = pB + kg;
        rb0 = *reinterpret_cast<const U128*>(pb);
        rb1 = *reinterpret_cast<const U128*>(pb + 8);
    };
    auto storeTile = [&](int buf) {
        *reinterpret_cast<U128*>(&lA[buf][rA * TK + kA])     = ra0;
        *reinterpret_cast<U128*>(&lA[buf][rA * TK + kA + 8]) = ra1;
        *reinterpret_cast<U128*>(&lB[buf][rA * TK + kA])     = rb0;
        *reinterpret_cast<U128*>(&lB[buf][rA * TK + kA + 8]) = rb1;
    };

    loadTile(0);
    storeTile(0);
    __syncthreads();

    const int nK = K / TK;
    for (int kt = 0; kt < nK; ++kt) {
        const int buf = kt & 1;
        if (kt + 1 < nK) loadTile((kt + 1) * TK);   // overlap with WMMAs below

        // A fragment (ISA wave32 16-bit 16x32 layout)
        Frag16 af;
        const int klo = half * 8;   // lanes 0-15: K 0-7/16-23; lanes 16-31: K 8-15/24-31
        af.q[0] = *reinterpret_cast<const U128*>(&lA[buf][(wm + l16) * TK + klo]);
        af.q[1] = *reinterpret_cast<const U128*>(&lA[buf][(wm + l16) * TK + klo + 16]);

        #pragma unroll
        for (int t = 0; t < NT; ++t) {
            Frag16 bfrag;
            const int nB = t * 16 + l16;
            const int kh = half * 8;    // lanes 0-15: K 0-7/16-23; lanes 16-31: K 8-15/24-31
            bfrag.q[0] = *reinterpret_cast<const U128*>(&lB[buf][nB * TK + kh]);
            bfrag.q[1] = *reinterpret_cast<const U128*>(&lB[buf][nB * TK + kh + 16]);
            acc[t] = __builtin_amdgcn_wmma_f32_16x16x32_bf16(
                false, af.v, false, bfrag.v, (short)0, acc[t], false, false);
        }
        if (kt + 1 < nK) {
            __syncthreads();            // all waves done reading buf^1
            storeTile(buf ^ 1);
            __syncthreads();            // buf^1 ready
        }
    }

    // epilogue (C/D layout: VGPR r -> row r (lanes 0-15) / r+8 (lanes 16-31))
    #pragma unroll
    for (int t = 0; t < NT; ++t) {
        const int col = t * 16 + l16;
        const float bv = bias[col];
        #pragma unroll
        for (int r = 0; r < 8; ++r) {
            const int row = m0 + wm + r + half * 8;
            if (row < M) {
                float y = acc[t][r] + bv;
                const size_t o = (size_t)row * H + col;
                if (EPI == 0) {
                    C16[o] = f32_bf16(fmaxf(y, 0.0f));
                } else if (EPI == 1) {
                    C[o] = y;
                } else if (EPI == 2) {
                    float v = C[o] + y;                 // edge_f residual (fp32 master)
                    C[o] = v; C16[o] = f32_bf16(v);
                    atomicAdd(C2 + (size_t)dstIdx[row] * H + col, y);   // segment_sum
                } else if (EPI == 3) {
                    float v = C[o] + y;                 // node_f residual
                    C[o] = v; C16[o] = f32_bf16(v);
                } else {
                    C[o] = y; C16[o] = f32_bf16(y);     // dual linear
                }
            }
        }
    }
}

// ---------------------------------------------------------------------------
// Helper kernels
// ---------------------------------------------------------------------------
__global__ void convert_weight(const float* __restrict__ W,
                               unsigned short* __restrict__ WT, int K) {
    int i = blockIdx.x * blockDim.x + threadIdx.x;
    if (i >= K * H) return;
    int n = i / K, k = i - n * K;            // WT[n][k] = bf16(W[k][n])
    WT[i] = f32_bf16(W[(size_t)k * H + n]);
}

__global__ void zero_f32(float* p, long long n) {
    long long i = (long long)blockIdx.x * blockDim.x + threadIdx.x;
    if (i < n) p[i] = 0.0f;
}

__global__ void cvt_bf16_arr(const float* __restrict__ in,
                             unsigned short* __restrict__ out, long long n) {
    long long i = (long long)blockIdx.x * blockDim.x + threadIdx.x;
    if (i < n) out[i] = f32_bf16(in[i]);
}

__global__ void node_enc_l1(const float* __restrict__ x, const float* __restrict__ pos,
                            const float* __restrict__ W, const float* __restrict__ b,
                            unsigned short* __restrict__ out, int M) {
    int i = blockIdx.x * blockDim.x + threadIdx.x;
    if (i >= M * H) return;
    int n = i >> 7, h = i & (H - 1);
    float acc = b[h];
    #pragma unroll
    for (int k = 0; k < 13; ++k) acc += x[n * 13 + k] * W[k * H + h];
    acc += pos[n * 2 + 0] * W[13 * H + h] + pos[n * 2 + 1] * W[14 * H + h];
    out[i] = f32_bf16(fmaxf(acc, 0.0f));
}

__global__ void edge_enc_l1(const float* __restrict__ ea,
                            const float* __restrict__ W, const float* __restrict__ b,
                            unsigned short* __restrict__ out, int M) {
    int i = blockIdx.x * blockDim.x + threadIdx.x;
    if (i >= M * H) return;
    int e = i >> 7, h = i & (H - 1);
    float acc = b[h];
    #pragma unroll
    for (int k = 0; k < 3; ++k) acc += ea[e * 3 + k] * W[k * H + h];
    out[i] = f32_bf16(fmaxf(acc, 0.0f));
}

__global__ void bn_stats(const float* __restrict__ h, float* __restrict__ mean,
                         float* __restrict__ rstd, int M) {
    const int col = blockIdx.x;
    float s = 0.f, s2 = 0.f;
    for (int r = threadIdx.x; r < M; r += blockDim.x) {
        float v = h[(size_t)r * H + col];
        s += v; s2 += v * v;
    }
    __shared__ float sh[256], sh2[256];
    sh[threadIdx.x] = s; sh2[threadIdx.x] = s2;
    __syncthreads();
    for (int o = 128; o > 0; o >>= 1) {
        if ((int)threadIdx.x < o) { sh[threadIdx.x] += sh[threadIdx.x + o];
                                    sh2[threadIdx.x] += sh2[threadIdx.x + o]; }
        __syncthreads();
    }
    if (threadIdx.x == 0) {
        float mu = sh[0] / (float)M;
        float var = sh2[0] / (float)M - mu * mu;
        mean[col] = mu;
        rstd[col] = rsqrtf(var + 1e-5f);
    }
}

__global__ void bn_apply_relu(const float* __restrict__ h, const float* __restrict__ mean,
                              const float* __restrict__ rstd, const float* __restrict__ g,
                              const float* __restrict__ be, float* __restrict__ outf,
                              unsigned short* __restrict__ outh, long long n) {
    long long i = (long long)blockIdx.x * blockDim.x + threadIdx.x;
    if (i >= n) return;
    int c = (int)(i & (H - 1));
    float v = fmaxf((h[i] - mean[c]) * rstd[c] * g[c] + be[c], 0.0f);
    outf[i] = v;
    outh[i] = f32_bf16(v);
}

__global__ void final_linear(const float* __restrict__ h, const float* __restrict__ W,
                             const float* __restrict__ b, float* __restrict__ out, int M) {
    int i = blockIdx.x * blockDim.x + threadIdx.x;
    if (i >= M * 2) return;
    int n = i >> 1, c = i & 1;
    float acc = b[c];
    #pragma unroll 8
    for (int k = 0; k < H; ++k) acc += h[(size_t)n * H + k] * W[k * 2 + c];
    out[i] = acc;
}

// ---------------------------------------------------------------------------
extern "C" void kernel_launch(void* const* d_in, const int* in_sizes, int n_in,
                              void* d_out, int out_size, void* d_ws, size_t ws_size,
                              hipStream_t stream) {
    const int Nn = in_sizes[0] / 13;
    const int Ne = in_sizes[2] / 3;

    const float* x   = (const float*)d_in[0];
    const float* pos = (const float*)d_in[1];
    const float* ea  = (const float*)d_in[2];
    const int*   ei  = (const int*)d_in[3];
    const int* srcI = ei;            // edge_index[0]
    const int* dstI = ei + Ne;       // edge_index[1]

    int idx = 4;
    const float *niW[3], *niB[3], *eiW[3], *eiB[3];
    for (int l = 0; l < 3; ++l) { niW[l] = (const float*)d_in[idx++]; niB[l] = (const float*)d_in[idx++]; }
    for (int l = 0; l < 3; ++l) { eiW[l] = (const float*)d_in[idx++]; eiB[l] = (const float*)d_in[idx++]; }
    const float *meW[10][3], *meB[10][3], *mnW[10][3], *mnB[10][3];
    for (int i = 0; i < 10; ++i)
        for (int l = 0; l < 3; ++l) { meW[i][l] = (const float*)d_in[idx++]; meB[i][l] = (const float*)d_in[idx++]; }
    for (int i = 0; i < 10; ++i)
        for (int l = 0; l < 3; ++l) { mnW[i][l] = (const float*)d_in[idx++]; mnB[i][l] = (const float*)d_in[idx++]; }
    const float *oW[3], *oB[3];
    for (int l = 0; l < 3; ++l) { oW[l] = (const float*)d_in[idx++]; oB[l] = (const float*)d_in[idx++]; }
    const float *bng[2], *bnb[2];
    for (int l = 0; l < 2; ++l) { bng[l] = (const float*)d_in[idx++]; bnb[l] = (const float*)d_in[idx++]; }

    // ---- workspace carve ----
    char* wp = (char*)d_ws;
    auto carve = [&](size_t bytes) { void* r = (void*)wp; wp += (bytes + 255) & ~(size_t)255; return r; };
    float*          nf    = (float*)carve((size_t)Nn * H * 4);          // fp32 master
    unsigned short* nf16  = (unsigned short*)carve((size_t)Nn * H * 2); // bf16 mirror
    float*          ef    = (float*)carve((size_t)Ne * H * 4);
    unsigned short* ef16  = (unsigned short*)carve((size_t)Ne * H * 2);
    unsigned short* te0   = (unsigned short*)carve((size_t)Ne * H * 2);
    unsigned short* te1   = (unsigned short*)carve((size_t)Ne * H * 2);
    float*          aggr  = (float*)carve((size_t)Nn * H * 4);
    unsigned short* agg16 = (unsigned short*)carve((size_t)Nn * H * 2);
    unsigned short* tn0   = (unsigned short*)carve((size_t)Nn * H * 2);
    unsigned short* tn1   = (unsigned short*)carve((size_t)Nn * H * 2);
    float*          dh0   = (float*)carve((size_t)Nn * H * 4);          // decoder fp32 (BN input)
    float*          dh1   = (float*)carve((size_t)Nn * H * 4);
    unsigned short* dhb   = (unsigned short*)carve((size_t)Nn * H * 2);
    float* bmean = (float*)carve(H * 4);
    float* brstd = (float*)carve(H * 4);
    unsigned short* w_ni2 = (unsigned short*)carve((size_t)H * H * 2);
    unsigned short* w_ni3 = (unsigned short*)carve((size_t)H * H * 2);
    unsigned short* w_ei2 = (unsigned short*)carve((size_t)H * H * 2);
    unsigned short* w_ei3 = (unsigned short*)carve((size_t)H * H * 2);
    unsigned short *w_me[10][3], *w_mn[10][3];
    for (int i = 0; i < 10; ++i) {
        w_me[i][0] = (unsigned short*)carve((size_t)3 * H * H * 2);
        w_me[i][1] = (unsigned short*)carve((size_t)H * H * 2);
        w_me[i][2] = (unsigned short*)carve((size_t)H * H * 2);
        w_mn[i][0] = (unsigned short*)carve((size_t)2 * H * H * 2);
        w_mn[i][1] = (unsigned short*)carve((size_t)H * H * 2);
        w_mn[i][2] = (unsigned short*)carve((size_t)H * H * 2);
    }
    unsigned short* w_o1 = (unsigned short*)carve((size_t)H * H * 2);
    unsigned short* w_o2 = (unsigned short*)carve((size_t)H * H * 2);

    auto cvt = [&](const float* W, unsigned short* WT, int K) {
        int total = K * H;
        convert_weight<<<dim3((total + 255) / 256), 256, 0, stream>>>(W, WT, K);
    };
    cvt(niW[1], w_ni2, H); cvt(niW[2], w_ni3, H);
    cvt(eiW[1], w_ei2, H); cvt(eiW[2], w_ei3, H);
    for (int i = 0; i < 10; ++i) {
        cvt(meW[i][0], w_me[i][0], 3 * H); cvt(meW[i][1], w_me[i][1], H); cvt(meW[i][2], w_me[i][2], H);
        cvt(mnW[i][0], w_mn[i][0], 2 * H); cvt(mnW[i][1], w_mn[i][1], H); cvt(mnW[i][2], w_mn[i][2], H);
    }
    cvt(oW[0], w_o1, H); cvt(oW[1], w_o2, H);

    const unsigned short* nul16 = nullptr;
#define GEMM(MODE, EPI, M, K, A0, A1, WT, BIAS, C, C2, C16)                      \
    gemm128_wmma_bf16<MODE, EPI><<<dim3(((M) + TM - 1) / TM), 256, 0, stream>>>( \
        (M), (K), (A0), (A1), srcI, dstI, (WT), (BIAS), (C), (C2), (C16))

    // ---- encoders ----
    node_enc_l1<<<dim3((Nn * H + 255) / 256), 256, 0, stream>>>(x, pos, niW[0], niB[0], tn0, Nn);
    GEMM(0, 0, Nn, H, tn0, nul16, w_ni2, niB[1], (float*)nullptr, (float*)nullptr, tn1);
    GEMM(0, 4, Nn, H, tn1, nul16, w_ni3, niB[2], nf, (float*)nullptr, nf16);

    edge_enc_l1<<<dim3((Ne * H + 255) / 256), 256, 0, stream>>>(ea, eiW[0], eiB[0], te0, Ne);
    GEMM(0, 0, Ne, H, te0, nul16, w_ei2, eiB[1], (float*)nullptr, (float*)nullptr, te1);
    GEMM(0, 4, Ne, H, te1, nul16, w_ei3, eiB[2], ef, (float*)nullptr, ef16);

    // ---- message passing ----
    const long long nNH = (long long)Nn * H;
    for (int i = 0; i < 10; ++i) {
        zero_f32<<<dim3((unsigned)((nNH + 255) / 256)), 256, 0, stream>>>(aggr, nNH);
        // edge MLP: cat(nf[dst], nf[src], ef) -> relu -> relu -> linear(+res, +scatter-sum)
        GEMM(1, 0, Ne, 3 * H, nf16, ef16, w_me[i][0], meB[i][0], (float*)nullptr, (float*)nullptr, te0);
        GEMM(0, 0, Ne, H,     te0,  nul16, w_me[i][1], meB[i][1], (float*)nullptr, (float*)nullptr, te1);
        GEMM(0, 2, Ne, H,     te1,  nul16, w_me[i][2], meB[i][2], ef, aggr, ef16);
        cvt_bf16_arr<<<dim3((unsigned)((nNH + 255) / 256)), 256, 0, stream>>>(aggr, agg16, nNH);
        // node MLP: cat(nf, aggr) -> relu -> relu -> linear(+residual)
        GEMM(2, 0, Nn, 2 * H, nf16, agg16, w_mn[i][0], mnB[i][0], (float*)nullptr, (float*)nullptr, tn0);
        GEMM(0, 0, Nn, H,     tn0,  nul16, w_mn[i][1], mnB[i][1], (float*)nullptr, (float*)nullptr, tn1);
        GEMM(0, 3, Nn, H,     tn1,  nul16, w_mn[i][2], mnB[i][2], nf, (float*)nullptr, nf16);
    }

    // ---- decoder with training-mode batch-norm ----
    GEMM(0, 1, Nn, H, nf16, nul16, w_o1, oB[0], dh0, (float*)nullptr, (unsigned short*)nullptr);
    bn_stats<<<dim3(H), 256, 0, stream>>>(dh0, bmean, brstd, Nn);
    bn_apply_relu<<<dim3((unsigned)((nNH + 255) / 256)), 256, 0, stream>>>(dh0, bmean, brstd, bng[0], bnb[0], dh1, dhb, nNH);
    GEMM(0, 1, Nn, H, dhb, nul16, w_o2, oB[1], dh0, (float*)nullptr, (unsigned short*)nullptr);
    bn_stats<<<dim3(H), 256, 0, stream>>>(dh0, bmean, brstd, Nn);
    bn_apply_relu<<<dim3((unsigned)((nNH + 255) / 256)), 256, 0, stream>>>(dh0, bmean, brstd, bng[1], bnb[1], dh1, dhb, nNH);
    final_linear<<<dim3((Nn * 2 + 255) / 256), 256, 0, stream>>>(dh1, oW[2], oB[2], (float*)d_out, Nn);

#undef GEMM
    (void)n_in; (void)out_size; (void)ws_size;
}
